// PAM_53377853554971
// MI455X (gfx1250) — matrix operand order repaired and checked by reading, not compiled, not run on hardware
//
#include <hip/hip_runtime.h>

#define BATCH 4
#define CDIM  256
#define NDIM  4096
#define BN_EPS 1e-5f

typedef __attribute__((ext_vector_type(16))) __bf16 v16bf;
typedef __attribute__((ext_vector_type(8)))  __bf16 v8bf;
typedef __attribute__((ext_vector_type(8)))  float  v8f;

static __device__ __forceinline__ v16bf combine16(v8bf lo, v8bf hi) {
    v16bf r;
#pragma unroll
    for (int i = 0; i < 8; ++i) { r[i] = lo[i]; r[i + 8] = hi[i]; }
    return r;
}

// ---------------------------------------------------------------------------
// Kernel 1: fused projections  a = Wa@x+ba, b = Wb@x+bb, m = BN(Wm@x+bm)
// Outputs (bf16):  aT [B][N][C], bT [B][N][C]  (row-major over K=c, A/B-tile
// friendly),  Mm [B][C][N]  (row-major over K=m for the second GEMM B-tiles).
// One wave per 16x16 output tile; shared x B-operand feeds all three GEMMs.
// All operands are L2-resident (192 MB L2), so per-o-tile x re-reads are cheap.
// ---------------------------------------------------------------------------
__global__ __launch_bounds__(32)
void pam_proj_kernel(const float* __restrict__ x,
                     const float* __restrict__ Wa, const float* __restrict__ ba,
                     const float* __restrict__ Wb, const float* __restrict__ bb,
                     const float* __restrict__ Wm, const float* __restrict__ bm,
                     const float* __restrict__ gamma, const float* __restrict__ beta,
                     const float* __restrict__ rmean, const float* __restrict__ rvar,
                     __bf16* __restrict__ aT, __bf16* __restrict__ bT,
                     __bf16* __restrict__ Mm)
{
    const int lane = threadIdx.x & 31;
    const int half = lane >> 4;
    const int l16  = lane & 15;
    const int n0 = blockIdx.x * 16;
    const int o0 = blockIdx.y * 16;
    const int b  = blockIdx.z;

    const int orow = o0 + l16;        // A-operand row (lane -> M)
    v8f acc_a = {}, acc_b = {}, acc_m = {};

#pragma unroll
    for (int kt = 0; kt < 8; ++kt) {
        const int c0 = kt * 32;
        // A-operand: per-lane K = {half*8+0..7, 16+half*8+0..7}
        v16bf Aa, Ab, Am;
#pragma unroll
        for (int i = 0; i < 8; ++i) {
            const int cLo = c0 + half * 8 + i;
            const int cHi = c0 + 16 + half * 8 + i;
            Aa[i]     = (__bf16)Wa[orow * CDIM + cLo];
            Aa[i + 8] = (__bf16)Wa[orow * CDIM + cHi];
            Ab[i]     = (__bf16)Wb[orow * CDIM + cLo];
            Ab[i + 8] = (__bf16)Wb[orow * CDIM + cHi];
            Am[i]     = (__bf16)Wm[orow * CDIM + cLo];
            Am[i + 8] = (__bf16)Wm[orow * CDIM + cHi];
        }
        // B-operand from x[b][c][n]: lane -> column n, K = {half*8+0..7, 16+half*8+0..7}
        v16bf Bx;
#pragma unroll
        for (int i = 0; i < 16; ++i) {
            const int c = c0 + (i < 8 ? half * 8 + i : 16 + half * 8 + (i - 8));
            Bx[i] = (__bf16)x[((size_t)b * CDIM + c) * NDIM + n0 + l16];
        }
        acc_a = __builtin_amdgcn_wmma_f32_16x16x32_bf16(false, Aa, false, Bx, (short)0, acc_a, false, false);
        acc_b = __builtin_amdgcn_wmma_f32_16x16x32_bf16(false, Ab, false, Bx, (short)0, acc_b, false, false);
        acc_m = __builtin_amdgcn_wmma_f32_16x16x32_bf16(false, Am, false, Bx, (short)0, acc_m, false, false);
    }

    // Epilogue: C/D layout -> VGPR r holds (M = r + 8*half, N = l16)
#pragma unroll
    for (int r = 0; r < 8; ++r) {
        const int o = o0 + r + half * 8;
        const int n = n0 + l16;
        const float av  = acc_a[r] + ba[o];
        const float bv  = acc_b[r] + bb[o];
        const float inv = rsqrtf(rvar[o] + BN_EPS);
        const float s   = inv * gamma[o];
        const float t   = (bm[o] - rmean[o]) * s + beta[o];
        const float mv  = acc_m[r] * s + t;
        aT[((size_t)b * NDIM + n) * CDIM + o] = (__bf16)av;
        bT[((size_t)b * NDIM + n) * CDIM + o] = (__bf16)bv;
        Mm[((size_t)b * CDIM + o) * NDIM + n] = (__bf16)mv;
    }
}

// ---------------------------------------------------------------------------
// Kernel 2: fused flash-attention  out = feature + alpha * softmax(a^T b) m^T
// One wave owns 16 query rows; full C=256 output in 16 f32 WMMA accum tiles.
// 64 keys/iter (4 S-tiles): the per-iteration VALU costs (O-rescale 128 muls,
// row max/sum shuffle trees) amortize over twice the keys vs a 32-key block,
// while WMMA work per key is unchanged (64 WMMAs / 64 keys / iter).
// ---------------------------------------------------------------------------
__global__ __launch_bounds__(128)
void pam_attn_kernel(const float* __restrict__ feature, const float* __restrict__ alpha,
                     const __bf16* __restrict__ aT, const __bf16* __restrict__ bT,
                     const __bf16* __restrict__ Mm, float* __restrict__ out)
{
    __shared__ __bf16 ldsP[4][16 * 64 + 32];   // per-wave P staging (+pad)

    const int lane = threadIdx.x & 31;
    const int wave = threadIdx.x >> 5;
    const int half = lane >> 4;
    const int l16  = lane & 15;
    const int b  = blockIdx.y;
    const int n0 = (blockIdx.x * 4 + wave) * 16;

    // Preload query A-operand tiles (8 x 16x32 bf16, resident for whole loop)
    v16bf Aq[8];
#pragma unroll
    for (int kt = 0; kt < 8; ++kt) {
        const __bf16* p = aT + ((size_t)b * NDIM + (n0 + l16)) * CDIM + kt * 32;
        v8bf lo = *(const v8bf*)(p + half * 16);
        v8bf hi = *(const v8bf*)(p + half * 16 + 8);
        Aq[kt] = combine16(lo, hi);
    }

    v8f O[16];
#pragma unroll
    for (int t = 0; t < 16; ++t) O[t] = (v8f){};
    float mrow[8], lrow[8];
#pragma unroll
    for (int r = 0; r < 8; ++r) { mrow[r] = -3.0e38f; lrow[r] = 0.0f; }

    __bf16* myP = &ldsP[wave][0];

    for (int km = 0; km < NDIM; km += 64) {
        // ---- logits: S[16x64] = a_rows (16xC) . b_cols^T  (4 16x16 tiles) ----
        v8f S[4];
#pragma unroll
        for (int j = 0; j < 4; ++j) S[j] = (v8f){};
#pragma unroll
        for (int kt = 0; kt < 8; ++kt) {
#pragma unroll
            for (int j = 0; j < 4; ++j) {
                const __bf16* pb =
                    bT + ((size_t)b * NDIM + (km + 16 * j + l16)) * CDIM + kt * 32 + half * 16;
                v16bf Bt = *(const v16bf*)pb;
                S[j] = __builtin_amdgcn_wmma_f32_16x16x32_bf16(false, Aq[kt], false, Bt,
                                                               (short)0, S[j], false, false);
            }
        }
        if (km + 64 < NDIM) {
            __builtin_prefetch(bT + ((size_t)b * NDIM + (km + 64 + l16)) * CDIM, 0, 1);
            __builtin_prefetch(Mm + ((size_t)b * CDIM + l16) * NDIM + km + 64, 0, 1);
        }

        // ---- online softmax over the 64 new columns ----
        float scale[8];
#pragma unroll
        for (int r = 0; r < 8; ++r) {
            float t = fmaxf(fmaxf(S[0][r], S[1][r]), fmaxf(S[2][r], S[3][r]));
            t = fmaxf(t, __shfl_xor(t, 1, 32));
            t = fmaxf(t, __shfl_xor(t, 2, 32));
            t = fmaxf(t, __shfl_xor(t, 4, 32));
            t = fmaxf(t, __shfl_xor(t, 8, 32));
            const float mnew = fmaxf(mrow[r], t);
            float rs = 0.0f;
#pragma unroll
            for (int j = 0; j < 4; ++j) {
                const float p = __expf(S[j][r] - mnew);
                S[j][r] = p;
                rs += p;
            }
            rs += __shfl_xor(rs, 1, 32);
            rs += __shfl_xor(rs, 2, 32);
            rs += __shfl_xor(rs, 4, 32);
            rs += __shfl_xor(rs, 8, 32);
            const float sc = __expf(mrow[r] - mnew);
            lrow[r] = lrow[r] * sc + rs;
            mrow[r] = mnew;
            scale[r] = sc;
        }
#pragma unroll
        for (int t = 0; t < 16; ++t) {
#pragma unroll
            for (int r = 0; r < 8; ++r) O[t][r] *= scale[r];
        }

        // ---- S (C/D layout) -> P (A-operand layout) via per-wave LDS ----
#pragma unroll
        for (int r = 0; r < 8; ++r) {
            const int row = r + half * 8;
#pragma unroll
            for (int j = 0; j < 4; ++j)
                myP[row * 64 + 16 * j + l16] = (__bf16)S[j][r];
        }
        asm volatile("s_wait_dscnt 0" ::: "memory");
        v16bf Pa0, Pa1;
        {
            const __bf16* p = myP + l16 * 64;
            Pa0 = combine16(*(const v8bf*)(p + half * 16),
                            *(const v8bf*)(p + half * 16 + 8));
            Pa1 = combine16(*(const v8bf*)(p + 32 + half * 16),
                            *(const v8bf*)(p + 40 + half * 16));
        }

        // ---- O[16xC] += P(16x64) x M(64x16) per output-channel tile ----
#pragma unroll
        for (int co = 0; co < 16; ++co) {
            const __bf16* pm = Mm + ((size_t)b * CDIM + co * 16 + l16) * NDIM + km + half * 16;
            v16bf Bm0 = *(const v16bf*)pm;
            v16bf Bm1 = *(const v16bf*)(pm + 32);
            O[co] = __builtin_amdgcn_wmma_f32_16x16x32_bf16(false, Pa0, false, Bm0,
                                                            (short)0, O[co], false, false);
            O[co] = __builtin_amdgcn_wmma_f32_16x16x32_bf16(false, Pa1, false, Bm1,
                                                            (short)0, O[co], false, false);
        }
    }

    // ---- epilogue: out[b][c][n] = feature + alpha * O[n][c] / l[n] ----
    const float al = alpha[0];
    float rinv[8];
#pragma unroll
    for (int r = 0; r < 8; ++r) rinv[r] = 1.0f / lrow[r];
#pragma unroll
    for (int co = 0; co < 16; ++co) {
#pragma unroll
        for (int r = 0; r < 8; ++r) {
            const int n = n0 + r + half * 8;
            const int c = co * 16 + l16;
            const size_t idx = ((size_t)b * CDIM + c) * NDIM + n;
            out[idx] = feature[idx] + al * (O[co][r] * rinv[r]);
        }
    }
}

// ---------------------------------------------------------------------------
extern "C" void kernel_launch(void* const* d_in, const int* in_sizes, int n_in,
                              void* d_out, int out_size, void* d_ws, size_t ws_size,
                              hipStream_t stream) {
    const float* feature = (const float*)d_in[0];
    const float* Wa   = (const float*)d_in[1];
    const float* ba   = (const float*)d_in[2];
    const float* Wb   = (const float*)d_in[3];
    const float* bb   = (const float*)d_in[4];
    const float* Wm   = (const float*)d_in[5];
    const float* bm   = (const float*)d_in[6];
    const float* gamm = (const float*)d_in[7];
    const float* beta = (const float*)d_in[8];
    const float* rmean = (const float*)d_in[9];
    const float* rvar  = (const float*)d_in[10];
    const float* alpha = (const float*)d_in[11];
    float* out = (float*)d_out;

    __bf16* aT = (__bf16*)d_ws;
    __bf16* bT = aT + (size_t)BATCH * NDIM * CDIM;
    __bf16* Mm = bT + (size_t)BATCH * NDIM * CDIM;

    dim3 g1(NDIM / 16, CDIM / 16, BATCH);
    pam_proj_kernel<<<g1, 32, 0, stream>>>(feature, Wa, ba, Wb, bb, Wm, bm,
                                           gamm, beta, rmean, rvar, aT, bT, Mm);

    dim3 g2(NDIM / 16 / 4, BATCH);
    pam_attn_kernel<<<g2, 128, 0, stream>>>(feature, alpha, aT, bT, Mm, out);
}
